// GaussianKernel_65223373357286
// MI455X (gfx1250) — compile-verified, matrix operand order checked by reading, not run on hardware
//
#include <hip/hip_runtime.h>

typedef __attribute__((ext_vector_type(16))) _Float16 v16h;
typedef __attribute__((ext_vector_type(8)))  _Float16 v8h;
typedef __attribute__((ext_vector_type(8)))  float    v8f;

#define GN 8192
#define GC 512
#define GK 64

// ---------------- Phase 1: phi[n,k] = exp(-||(f[n]-mu[k])/sigma[k]||^2) (*w[k]) ----
// Block = 256 threads: 4 rows (n) x 64 centroids (k). f[n,c] is uniform across each
// wave (broadcast load); means/scales are 256KB total -> L2 resident.
__global__ void phi_kernel(const float* __restrict__ f,
                           const float* __restrict__ means,
                           const float* __restrict__ scales,
                           const float* __restrict__ weights,
                           _Float16* __restrict__ phi,
                           int use_w) {
  const int t = threadIdx.x;
  const int k = t & (GK - 1);
  const int r = t >> 6;
  const int n = blockIdx.x * 4 + r;

  const float4* fr = (const float4*)(f      + (size_t)n * GC);
  const float4* mr = (const float4*)(means  + (size_t)k * GC);
  const float4* sr = (const float4*)(scales + (size_t)k * GC);

  float acc = 0.f;
#pragma unroll 4
  for (int c = 0; c < GC / 4; ++c) {
    float4 fv = fr[c];
    float4 mv = mr[c];
    float4 sv = sr[c];
    float d0 = (fv.x - mv.x) / sv.x;
    float d1 = (fv.y - mv.y) / sv.y;
    float d2 = (fv.z - mv.z) / sv.z;
    float d3 = (fv.w - mv.w) / sv.w;
    acc += d0 * d0 + d1 * d1 + d2 * d2 + d3 * d3;
  }
  float p = __expf(-acc);
  if (use_w) p *= weights[k];
  phi[(size_t)n * GK + k] = (_Float16)p;
}

// ---------------- Phase 2: out = phi_i @ phi_j^T via v_wmma_f32_16x16x32_f16 ------
// 16-bit A-matrix 16x32 layout (wave32): lane L holds row (L&15),
// K = {(L>>4)*8 .. +7} in v[0..7] halves and {.. +16..+23} in v[8..15] halves.
// B fragment for B = phi_j^T is the identical row-major load from phi_j.
__device__ __forceinline__ v16h load_frag(const _Float16* __restrict__ p,
                                          int row0, int kbase, int lane) {
  const _Float16* q = p + (size_t)(row0 + (lane & 15)) * GK + kbase + ((lane >> 4) << 3);
  v8h lo = *(const v8h*)q;         // global_load_b128
  v8h hi = *(const v8h*)(q + 16);  // global_load_b128
  v16h r;
#pragma unroll
  for (int i = 0; i < 8; ++i) { r[i] = lo[i]; r[i + 8] = hi[i]; }
  return r;
}

__global__ void gemm_kernel(const _Float16* __restrict__ phi_i,
                            const _Float16* __restrict__ phi_j,
                            float* __restrict__ out) {
  const int lane = threadIdx.x & 31;
  const int w    = threadIdx.x >> 5;
  const int i0    = blockIdx.x * 16;                 // output row tile
  const int jbase = (blockIdx.y * 8 + w) * 64;       // 4 col tiles per wave

  // A fragments reused across the 4 column tiles (K=64 -> two 16x16x32 steps).
  v16h a0 = load_frag(phi_i, i0, 0,  lane);
  v16h a1 = load_frag(phi_i, i0, 32, lane);

#pragma unroll
  for (int jt = 0; jt < 4; ++jt) {
    const int j0 = jbase + jt * 16;
    v16h b0 = load_frag(phi_j, j0, 0,  lane);
    v16h b1 = load_frag(phi_j, j0, 32, lane);

    v8f acc = {};
    acc = __builtin_amdgcn_wmma_f32_16x16x32_f16(false, a0, false, b0,
                                                 (short)0, acc, false, false);
    acc = __builtin_amdgcn_wmma_f32_16x16x32_f16(false, a1, false, b1,
                                                 (short)0, acc, false, false);

    // D layout: VGPR r -> row (r + 8*(lane>>4)), col (lane&15).
    // 256MB streaming output > 192MB L2 -> non-temporal stores.
    const int col   = j0 + (lane & 15);
    const int rbase = i0 + ((lane >> 4) << 3);
#pragma unroll
    for (int r = 0; r < 8; ++r) {
      __builtin_nontemporal_store(acc[r], &out[(size_t)(rbase + r) * GN + col]);
    }
  }
}

extern "C" void kernel_launch(void* const* d_in, const int* in_sizes, int n_in,
                              void* d_out, int out_size, void* d_ws, size_t ws_size,
                              hipStream_t stream) {
  const float* f_i     = (const float*)d_in[0];
  const float* f_j     = (const float*)d_in[1];
  const float* means   = (const float*)d_in[2];
  const float* scales  = (const float*)d_in[3];
  const float* weights = (const float*)d_in[4];
  float* out = (float*)d_out;

  _Float16* phi_i = (_Float16*)d_ws;                 // 1 MB
  _Float16* phi_j = phi_i + (size_t)GN * GK;         // 1 MB

  phi_kernel<<<dim3(GN / 4), 256, 0, stream>>>(f_i, means, scales, weights, phi_i, 1);
  phi_kernel<<<dim3(GN / 4), 256, 0, stream>>>(f_j, means, scales, weights, phi_j, 0);

  // 512 row-tiles x 16 y-blocks; each block = 8 waves x (16x64 strip) = 16x512.
  gemm_kernel<<<dim3(GN / 16, GN / 512), 256, 0, stream>>>(phi_i, phi_j, out);
}
